// GraNDAG_5145370820726
// MI455X (gfx1250) — matrix operand order repaired and checked by reading, not compiled, not run on hardware
//
#include <hip/hip_runtime.h>

typedef __attribute__((ext_vector_type(16))) _Float16 v16h;
typedef __attribute__((ext_vector_type(8)))  _Float16 v8h;
typedef __attribute__((ext_vector_type(8)))  float    v8f;

#define B_N 4096
#define V_N 64
#define H_N 512

// ---- Prep 1: x (f32, B x V) -> xh (f16, row-major) -------------------------
__global__ void cvt_x_f16(const float* __restrict__ x, _Float16* __restrict__ xh, int n) {
    int i = blockIdx.x * blockDim.x + threadIdx.x;
    if (i < n) xh[i] = (_Float16)x[i];
}

// ---- Prep 2: build WfullT[v][h][k] (f16), fusing sigmoid(adj), gather, mask -
// Wfull[v,k,h] = sigmoid(adj_weights[v,k]) * W1[v, j, h],  j = k - (k>v),
// row k==v zeroed (adj diagonal mask + idx gather folded away).
// Stored transposed [v][h][k] so the WMMA B-operand is one contiguous 32B load.
__global__ void build_wfullT(const float* __restrict__ adjw,
                             const float* __restrict__ W1,
                             _Float16* __restrict__ wfT) {
    int id = blockIdx.x * blockDim.x + threadIdx.x;   // < 64*512*64
    int k = id & 63;
    int h = (id >> 6) & 511;
    int v = id >> 15;
    float val = 0.0f;
    if (k != v) {
        float aw = adjw[v * V_N + k];
        float s  = 1.0f / (1.0f + expf(-aw));
        int  j   = k - (k > v ? 1 : 0);
        val = s * W1[((size_t)v * (V_N - 1) + j) * H_N + h];
    }
    wfT[id] = (_Float16)val;
}

// ---- Fused main kernel ------------------------------------------------------
// Per block: one v, 32 B-rows, full H=512.  8 waves, each owns M=32 x N=64:
//   h = relu(x(32x64) @ Wfull[v](64x512) + b1[v])   [16 x v_wmma_f32_16x16x32_f16]
//   out[b,v] = h . W2[v] + b2[v]                    [fused epilogue, h stays in VGPRs]
// Each B tile load feeds TWO WMMAs (two M-halves) -> half the L2 traffic per MMA.
__global__ void __launch_bounds__(256)
grandag_fused(const _Float16* __restrict__ xh,
              const _Float16* __restrict__ wfT,
              const float* __restrict__ b1,
              const float* __restrict__ W2,
              const float* __restrict__ b2,
              float* __restrict__ out) {
    __shared__ float red[8 * 32];

    const int tid  = threadIdx.x;
    const int wave = tid >> 5;
    const int lane = tid & 31;
    const int l16  = lane & 15;
    const bool hi  = lane >= 16;

    const int mbase = blockIdx.x * 32;   // B-row tile (32 rows)
    const int v     = blockIdx.y;        // which node
    const int nbase = wave * 64;         // this wave's H slice

    // Accumulators: row-half 0 (M = mbase..+15) and row-half 1 (M = mbase+16..+31)
    v8f c00 = {}, c01 = {}, c02 = {}, c03 = {};
    v8f c10 = {}, c11 = {}, c12 = {}, c13 = {};

    const _Float16* xrow0 = xh + (size_t)(mbase + l16) * V_N;        // A half 0
    const _Float16* xrow1 = xrow0 + (size_t)16 * V_N;                // A half 1
    const _Float16* wv    = wfT + (size_t)v * H_N * V_N;

#pragma unroll
    for (int kc = 0; kc < 2; ++kc) {
        // A operand (16x32 f16): lane L, M = L%16; K contiguous (matches B):
        //   lanes 0-15 : K = kc*32 + 0..15
        //   lanes 16-31: K = kc*32 + 16..31
        const int kbA = kc * 32 + (hi ? 16 : 0);
        v8h a0lo = *(const v8h*)(xrow0 + kbA);
        v8h a0hi = *(const v8h*)(xrow0 + kbA + 8);
        v16h a0  = __builtin_shufflevector(a0lo, a0hi,
                     0,1,2,3,4,5,6,7,8,9,10,11,12,13,14,15);
        v8h a1lo = *(const v8h*)(xrow1 + kbA);
        v8h a1hi = *(const v8h*)(xrow1 + kbA + 8);
        v16h a1  = __builtin_shufflevector(a1lo, a1hi,
                     0,1,2,3,4,5,6,7,8,9,10,11,12,13,14,15);

        // B operand (32x16 f16): lane L, N = nbase + t*16 + L%16;
        //   lanes 0-15 hold K = kc*32 + 0..15, lanes 16-31 hold K = kc*32 + 16..31.
        // WfullT is [h][k] so this is one contiguous 32B load per lane.
        const int kbB = kc * 32 + (hi ? 16 : 0);
        const _Float16* bb = wv + kbB;

        v16h b0 = *(const v16h*)(bb + (size_t)(nbase +  0 + l16) * V_N);
        v16h b1t = *(const v16h*)(bb + (size_t)(nbase + 16 + l16) * V_N);
        v16h b2t = *(const v16h*)(bb + (size_t)(nbase + 32 + l16) * V_N);
        v16h b3t = *(const v16h*)(bb + (size_t)(nbase + 48 + l16) * V_N);

        c00 = __builtin_amdgcn_wmma_f32_16x16x32_f16(false, a0, false, b0,  (short)0, c00, false, false);
        c10 = __builtin_amdgcn_wmma_f32_16x16x32_f16(false, a1, false, b0,  (short)0, c10, false, false);
        c01 = __builtin_amdgcn_wmma_f32_16x16x32_f16(false, a0, false, b1t, (short)0, c01, false, false);
        c11 = __builtin_amdgcn_wmma_f32_16x16x32_f16(false, a1, false, b1t, (short)0, c11, false, false);
        c02 = __builtin_amdgcn_wmma_f32_16x16x32_f16(false, a0, false, b2t, (short)0, c02, false, false);
        c12 = __builtin_amdgcn_wmma_f32_16x16x32_f16(false, a1, false, b2t, (short)0, c12, false, false);
        c03 = __builtin_amdgcn_wmma_f32_16x16x32_f16(false, a0, false, b3t, (short)0, c03, false, false);
        c13 = __builtin_amdgcn_wmma_f32_16x16x32_f16(false, a1, false, b3t, (short)0, c13, false, false);
    }

    // Epilogue: C layout -> lane holds column N = t*16 + l16; VGPR r = rows r (lanes<16) / r+8.
    float p0[8], p1[8];
#pragma unroll
    for (int r = 0; r < 8; ++r) { p0[r] = 0.0f; p1[r] = 0.0f; }

#pragma unroll
    for (int t = 0; t < 4; ++t) {
        const int hcol = nbase + t * 16 + l16;
        const float bias = b1[v * H_N + hcol];
        const float w2v  = W2[v * H_N + hcol];
        v8f ca = (t == 0) ? c00 : (t == 1) ? c01 : (t == 2) ? c02 : c03;
        v8f cb = (t == 0) ? c10 : (t == 1) ? c11 : (t == 2) ? c12 : c13;
#pragma unroll
        for (int r = 0; r < 8; ++r) {
            float h0 = fmaxf(ca[r] + bias, 0.0f);
            float h1 = fmaxf(cb[r] + bias, 0.0f);
            p0[r] = fmaf(h0, w2v, p0[r]);
            p1[r] = fmaf(h1, w2v, p1[r]);
        }
    }

    // Reduce over the 16 columns held within each 16-lane half (rows differ per half).
#pragma unroll
    for (int r = 0; r < 8; ++r) {
#pragma unroll
        for (int off = 1; off < 16; off <<= 1) {
            p0[r] += __shfl_xor(p0[r], off, 32);
            p1[r] += __shfl_xor(p1[r], off, 32);
        }
    }

    if (l16 == 0) {
        const int rbase = hi ? 8 : 0;
#pragma unroll
        for (int r = 0; r < 8; ++r) {
            red[wave * 32 + rbase + r]      = p0[r];   // rows mbase + rbase + r
            red[wave * 32 + 16 + rbase + r] = p1[r];   // rows mbase + 16 + rbase + r
        }
    }
    __syncthreads();

    // Combine the 8 waves' H-slices, add b2, write out[b, v].
    if (tid < 32) {
        float s = b2[v];
#pragma unroll
        for (int w = 0; w < 8; ++w) s += red[w * 32 + tid];
        out[(size_t)(mbase + tid) * V_N + v] = s;
    }
}

extern "C" void kernel_launch(void* const* d_in, const int* in_sizes, int n_in,
                              void* d_out, int out_size, void* d_ws, size_t ws_size,
                              hipStream_t stream) {
    const float* x    = (const float*)d_in[0];
    const float* adjw = (const float*)d_in[1];
    const float* W1   = (const float*)d_in[2];
    const float* b1   = (const float*)d_in[3];
    const float* W2   = (const float*)d_in[4];
    const float* b2   = (const float*)d_in[5];
    float* out = (float*)d_out;

    _Float16* xh  = (_Float16*)d_ws;                                    // 4096*64 f16 = 512 KB
    _Float16* wfT = (_Float16*)((char*)d_ws + (size_t)B_N * V_N * 2);   // 64*512*64 f16 = 4 MB

    cvt_x_f16<<<(B_N * V_N) / 256, 256, 0, stream>>>(x, xh, B_N * V_N);
    build_wfullT<<<(V_N * H_N * V_N) / 256, 256, 0, stream>>>(adjw, W1, wfT);

    dim3 grid(B_N / 32, V_N);
    grandag_fused<<<grid, 256, 0, stream>>>(xh, wfT, b1, W2, b2, out);
}
